// CrossAttention_463856468091
// MI455X (gfx1250) — matrix operand order repaired and checked by reading, not compiled, not run on hardware
//
#include <hip/hip_runtime.h>
#include <hip/hip_bf16.h>

// ---- shapes (fixed by the reference) ----
#define B_   2
#define SQ   2048
#define SK   2048
#define H_   16
#define HK_  4
#define D_   64
#define WAVES 4

// fold log2(e) into the QK scale and the mask bias: softmax in base-2 domain
#define LOG2E    1.44269504088896340736f
#define QK_SCALE (0.125f * LOG2E)        // 1/sqrt(64) * log2(e)
#define PAD_NEG  (-10000.0f * LOG2E)

// workspace layout (bytes)
#define KH_BYTES  ((size_t)B_ * HK_ * SK * D_ * 2)   // 2 MB  f16 K rows
#define VT_BYTES  ((size_t)B_ * HK_ * SK * D_ * 2)   // 2 MB  f16 V transposed (d-major)
#define WS_KH_OFF 0
#define WS_VT_OFF KH_BYTES
#define WS_PAD_OFF (KH_BYTES + VT_BYTES)             // B*SK f32 mask bias

typedef __attribute__((ext_vector_type(16))) _Float16 v16h;
typedef __attribute__((ext_vector_type(8)))  _Float16 v8h;
typedef __attribute__((ext_vector_type(8)))  float    v8f;
typedef __attribute__((ext_vector_type(4)))  float    v4f;

// 8 consecutive f32 -> 8 f16
__device__ static inline v8h ld8h(const float* p) {
    v4f x0 = *(const v4f*)p;
    v4f x1 = *(const v4f*)(p + 4);
    v8h r;
#pragma unroll
    for (int i = 0; i < 4; ++i) {
        r[i]     = (_Float16)x0[i];
        r[4 + i] = (_Float16)x1[i];
    }
    return r;
}

// 16 consecutive f32 -> 16 f16, scaled
__device__ static inline v16h ld16h_scaled(const float* p, float s) {
    v16h r;
#pragma unroll
    for (int i = 0; i < 16; i += 4) {
        v4f x = *(const v4f*)(p + i);
        r[i]   = (_Float16)(x[0] * s); r[i+1] = (_Float16)(x[1] * s);
        r[i+2] = (_Float16)(x[2] * s); r[i+3] = (_Float16)(x[3] * s);
    }
    return r;
}

__device__ static inline v16h cat8(v8h a, v8h b) {
    v16h r;
#pragma unroll
    for (int i = 0; i < 8; ++i) { r[i] = a[i]; r[8 + i] = b[i]; }
    return r;
}

#define WMMA_F16(A, Bm, C) \
    __builtin_amdgcn_wmma_f32_16x16x32_f16(false, (A), false, (Bm), (short)0, (C), false, false)

// ---------------------------------------------------------------------------
// Kernel 1: one-time KV repack.  One thread per (b, hk, s) key row.
//   Kh[b][hk][s][d]  f16 row-major        (A-fragment friendly)
//   Vt[b][hk][d][s]  f16 key-major rows   (B-fragment friendly, pre-transposed)
//   padf[b][s]       f32 mask bias in log2 domain
// Amortized over H/Hk * Sq/16 = 512 reuses per element.
// ---------------------------------------------------------------------------
__global__ __launch_bounds__(256)
void repack_kv(const float* __restrict__ kv, const unsigned char* __restrict__ mask,
               _Float16* __restrict__ Kh, _Float16* __restrict__ Vt,
               float* __restrict__ padf) {
    const int t  = blockIdx.x * 256 + threadIdx.x;   // 0 .. B*HK*SK-1
    const int s  = t & (SK - 1);
    const int hk = (t >> 11) & (HK_ - 1);
    const int b  = t >> 13;

    const float* ksrc = kv + ((size_t)(b * SK + s) * 2 * HK_ + hk) * D_;  // c=0
    const float* vsrc = ksrc + (size_t)HK_ * D_;                          // c=1
    _Float16* kdst = Kh + ((size_t)(b * HK_ + hk) * SK + s) * D_;
    _Float16* vdst = Vt + ((size_t)(b * HK_ + hk) * D_) * SK + s;

#pragma unroll
    for (int i = 0; i < D_; i += 8)
        *(v8h*)(kdst + i) = ld8h(ksrc + i);

#pragma unroll
    for (int i = 0; i < D_; i += 4) {
        v4f x = *(const v4f*)(vsrc + i);
        vdst[(size_t)(i + 0) * SK] = (_Float16)x[0];
        vdst[(size_t)(i + 1) * SK] = (_Float16)x[1];
        vdst[(size_t)(i + 2) * SK] = (_Float16)x[2];
        vdst[(size_t)(i + 3) * SK] = (_Float16)x[3];
    }

    if (hk == 0)
        padf[b * SK + s] = mask[(size_t)b * SK + s] ? 0.0f : PAD_NEG;
}

// ---------------------------------------------------------------------------
// Kernel 2: flash attention, one wave per 16-query tile (wave32-native).
// 64 keys per trip: one online-softmax update amortized over 16 WMMAs.
// ---------------------------------------------------------------------------
__global__ __launch_bounds__(WAVES * 32)
void CrossAttention_fa_wmma(const float* __restrict__ q,
                            const _Float16* __restrict__ Kh,
                            const _Float16* __restrict__ Vt,
                            const float* __restrict__ padf,
                            float* __restrict__ out) {
    const int lane = threadIdx.x & 31;
    const int wave = threadIdx.x >> 5;
    const int tile = blockIdx.x * WAVES + wave;  // 4096 tiles total
    const int tq   = tile & 127;                 // Sq/16 = 128 tiles per (b,h)
    const int bh   = tile >> 7;
    const int h    = bh & 15;
    const int b    = bh >> 4;
    const int hk   = h >> 2;                     // GQA: 4 query heads per kv head

    const int n      = lane & 15;   // query column (and key row within a tile)
    const int hi     = lane >> 4;
    const int off8   = hi * 8;      // A-layout contraction sub-offset
    const int doff16 = hi * 16;     // contiguous-k contraction sub-offset (Q and K frags)

    // ---- Q tile as two persistent 32x16 B-fragments (Q^T), scale*log2e folded ----
    const float* qrow = q + ((size_t)((b * SQ + tq * 16 + n) * H_) + h) * D_;
    const v16h bq0 = ld16h_scaled(qrow + doff16,      QK_SCALE);  // d 0..31
    const v16h bq1 = ld16h_scaled(qrow + 32 + doff16, QK_SCALE);  // d 32..63

    // ---- O (16q x 64d as 4 C-frags: row = query r+8*hi, col = d 16j+n) ----
    v8f acc[4];
#pragma unroll
    for (int j = 0; j < 4; ++j) acc[j] = (v8f){};
    // softmax state per lane: query column n (replicated across the two lane halves)
    float mstate = -1e30f, lstate = 0.0f;

    const _Float16* kbase = Kh + (size_t)(b * HK_ + hk) * SK * D_;
    const _Float16* vbase = Vt + (size_t)(b * HK_ + hk) * D_ * SK;
    const float*    prow  = padf + b * SK;

    for (int s0 = 0; s0 < SK; s0 += 64) {
        if (s0 + 64 < SK) {   // prefetch next K/V f16 tiles (global_prefetch_b8)
            __builtin_prefetch(kbase + (size_t)(s0 + 64 + lane) * D_, 0, 1);
            __builtin_prefetch(kbase + (size_t)(s0 + 96 + lane) * D_, 0, 1);
            __builtin_prefetch(vbase + (size_t)(lane * 2) * SK + s0 + 64, 0, 1);
        }

        // ---- 4 score tiles: S^T = K @ Q^T (rows = keys, cols = queries) ----
        v8f st[4];
#pragma unroll
        for (int t4 = 0; t4 < 4; ++t4) {
            const _Float16* kr = kbase + (size_t)(s0 + 16 * t4 + n) * D_;
            const v16h akl = *(const v16h*)(kr + doff16);
            const v16h akh = *(const v16h*)(kr + 32 + doff16);
            v8f s = (v8f){};
            s = WMMA_F16(akl, bq0, s);
            s = WMMA_F16(akh, bq1, s);
            // pre-expanded mask bias: key = s0 + 16*t4 + off8 + r
            const v4f pa4 = *(const v4f*)(prow + s0 + 16 * t4 + off8);
            const v4f pb4 = *(const v4f*)(prow + s0 + 16 * t4 + off8 + 4);
#pragma unroll
            for (int r = 0; r < 4; ++r) { s[r] += pa4[r]; s[4 + r] += pb4[r]; }
            st[t4] = s;
        }

        // ---- one online-softmax update per 64 keys ----
        float tmax = fmaxf(fmaxf(st[0][0], st[1][0]), fmaxf(st[2][0], st[3][0]));
#pragma unroll
        for (int r = 1; r < 8; ++r) {
            const float a01 = fmaxf(st[0][r], st[1][r]);
            const float a23 = fmaxf(st[2][r], st[3][r]);
            tmax = fmaxf(tmax, fmaxf(a01, a23));
        }
        tmax = fmaxf(tmax, __shfl_xor(tmax, 16));
        const float mn = fmaxf(mstate, tmax);

        float p[4][8];
        float rs = 0.0f;
#pragma unroll
        for (int t4 = 0; t4 < 4; ++t4)
#pragma unroll
            for (int r = 0; r < 8; ++r) {
                p[t4][r] = __builtin_amdgcn_exp2f(st[t4][r] - mn);
                rs += p[t4][r];
            }
        rs += __shfl_xor(rs, 16);
        const float al = __builtin_amdgcn_exp2f(mstate - mn);
        lstate = lstate * al + rs;
        mstate = mn;

        // ---- rescale O once per 64 keys (alpha broadcast column->row owners) ----
#pragma unroll
        for (int r = 0; r < 8; ++r) {
            const float ar = __shfl(al, r + off8, 32);
            acc[0][r] *= ar; acc[1][r] *= ar; acc[2][r] *= ar; acc[3][r] *= ar;
        }

        // ---- P: S^T C-layout == P A-layout, pure in-lane f32->f16 pack ----
        v16h pa01, pa23;   // keys [s0,s0+32) and [s0+32,s0+64)
#pragma unroll
        for (int r = 0; r < 8; ++r) {
            pa01[r]     = (_Float16)p[0][r];
            pa01[8 + r] = (_Float16)p[1][r];
            pa23[r]     = (_Float16)p[2][r];
            pa23[8 + r] = (_Float16)p[3][r];
        }

        // ---- O += P @ V : pre-transposed f16 V, two b128 loads per fragment ----
#pragma unroll
        for (int j = 0; j < 4; ++j) {
            const _Float16* vr = vbase + (size_t)(16 * j + n) * SK + s0 + off8;
            const v16h bv0 = cat8(*(const v8h*)vr,        *(const v8h*)(vr + 16));
            acc[j] = WMMA_F16(pa01, bv0, acc[j]);
            const v16h bv1 = cat8(*(const v8h*)(vr + 32), *(const v8h*)(vr + 48));
            acc[j] = WMMA_F16(pa23, bv1, acc[j]);
        }
    }

    // ---- epilogue: O / l (broadcast l from query-column owners), coalesced stores ----
    float rrec[8];
#pragma unroll
    for (int r = 0; r < 8; ++r)
        rrec[r] = 1.0f / __shfl(lstate, r + off8, 32);

#pragma unroll
    for (int j = 0; j < 4; ++j) {
#pragma unroll
        for (int r = 0; r < 8; ++r) {
            const int row = r + 8 * hi;
            const int tr  = tq * 16 + row;
            out[((size_t)((b * SQ + tr) * H_) + h) * D_ + 16 * j + n] = acc[j][r] * rrec[r];
        }
    }
}

extern "C" void kernel_launch(void* const* d_in, const int* in_sizes, int n_in,
                              void* d_out, int out_size, void* d_ws, size_t ws_size,
                              hipStream_t stream) {
    (void)in_sizes; (void)n_in; (void)ws_size; (void)out_size;
    const float*         q    = (const float*)d_in[0];
    const float*         kv   = (const float*)d_in[1];
    const unsigned char* mask = (const unsigned char*)d_in[2];   // jnp bool -> 1 byte
    float*               out  = (float*)d_out;

    _Float16* Kh   = (_Float16*)((char*)d_ws + WS_KH_OFF);
    _Float16* Vt   = (_Float16*)((char*)d_ws + WS_VT_OFF);
    float*    padf = (float*)   ((char*)d_ws + WS_PAD_OFF);

    // one-time repack: B*HK*SK = 16384 key rows, 256 threads/block
    repack_kv<<<(B_ * HK_ * SK) / 256, 256, 0, stream>>>(kv, mask, Kh, Vt, padf);

    const int tiles  = B_ * H_ * (SQ / 16);      // 4096 wave-tiles
    const int blocks = tiles / WAVES;            // 1024 blocks of 4 waves
    CrossAttention_fa_wmma<<<blocks, WAVES * 32, 0, stream>>>(q, Kh, Vt, padf, out);
}
